// SelfAttentionPooler_56813827392212
// MI455X (gfx1250) — matrix operand order repaired and checked by reading, not compiled, not run on hardware
//
#include <hip/hip_runtime.h>

// ---------------------------------------------------------------------------
// SelfAttentionPooler for MI455X (gfx1250, wave32, bf16 WMMA + async->LDS)
//   B=16, S=1024, D=768, DOUT=256
//   out  = softmax(mask((emb@Wq+bq)/sqrt(D) @ (emb@Wk+bk)^T)) @ (emb@Wv+bv)
//   d_out = [out (B*S*DOUT f32) | attn (B*S*S f32)]
// ---------------------------------------------------------------------------

#define BATCH 16
#define SEQ   1024
#define DIM   768
#define DOUTC 256

typedef __bf16 bf16_t;
typedef __attribute__((ext_vector_type(16))) __bf16 v16bf;
typedef __attribute__((ext_vector_type(8)))  __bf16 v8bf;
typedef __attribute__((ext_vector_type(8)))  float  v8f;
typedef __attribute__((ext_vector_type(4)))  float  v4f;
typedef __attribute__((ext_vector_type(4)))  int    v4i;

__device__ __constant__ const float INV_TEMP = 0.03608439182435161f; // 1/sqrt(768)

// Async global->LDS path (gfx1250): GLOBAL_LOAD_ASYNC_TO_LDS_B128 + ASYNCcnt.
// Builtin signature (from hipcc diagnostic): arg0 = v4i addrspace(1)* (global
// source), arg1 = v4i addrspace(3)* (LDS dest), then imm offset, imm cpol.
#if __has_builtin(__builtin_amdgcn_global_load_async_to_lds_b128) && \
    __has_builtin(__builtin_amdgcn_s_wait_asynccnt)
#define USE_ASYNC_LDS 1
typedef __attribute__((address_space(1))) v4i as1_v4i;
typedef __attribute__((address_space(3))) v4i as3_v4i;
#else
#define USE_ASYNC_LDS 0
#endif

// -------- WMMA fragment loaders (wave32, ISA 7.12.2 layouts) ----------------

// A-matrix 16x32 bf16 from an f32 row-major source (convert on the fly).
// lanes 0-15: row = lane,    K = {k0..k0+7, k0+16..k0+23}
// lanes 16-31: row = lane-16, K = {k0+8..k0+15, k0+24..k0+31}
__device__ __forceinline__ v16bf load_a_frag_f32(const float* __restrict__ base,
                                                 int ld, int row, int k0, int lane) {
    const int half = (lane >> 4) & 1;
    const int r = row + (lane & 15);
    const float* p = base + (size_t)r * ld + (k0 + half * 8);
    v4f c0 = *(const v4f*)(p + 0);
    v4f c1 = *(const v4f*)(p + 4);
    v4f c2 = *(const v4f*)(p + 16);
    v4f c3 = *(const v4f*)(p + 20);
    v16bf out;
#pragma unroll
    for (int i = 0; i < 4; ++i) {
        out[i]      = (__bf16)c0[i];
        out[4 + i]  = (__bf16)c1[i];
        out[8 + i]  = (__bf16)c2[i];
        out[12 + i] = (__bf16)c3[i];
    }
    return out;
}

// A-matrix 16x32 bf16 from a bf16 row-major source (global or LDS).
__device__ __forceinline__ v16bf load_a_frag_bf16(const bf16_t* base,
                                                  int ld, int row, int k0, int lane) {
    const int half = (lane >> 4) & 1;
    const int r = row + (lane & 15);
    const bf16_t* p = base + (size_t)r * ld + (k0 + half * 8);
    union { v16bf v; v8bf h[2]; } u;
    u.h[0] = *(const v8bf*)(p);
    u.h[1] = *(const v8bf*)(p + 16);
    return u.v;
}

// B-matrix 32x16 bf16. Source stored so that column n of B is a contiguous
// row: src[n*ld + k]. lanes 0-15: col=lane, K={k0..k0+7, k0+16..k0+23} ; lanes 16-31: K +8.
__device__ __forceinline__ v16bf load_b_frag(const bf16_t* __restrict__ base,
                                             int ld, int n0, int k0, int lane) {
    const int half = (lane >> 4) & 1;
    const int n = n0 + (lane & 15);
    const bf16_t* p = base + (size_t)n * ld + (k0 + half * 8);
    union { v16bf v; v8bf h[2]; } u;
    u.h[0] = *(const v8bf*)(p);
    u.h[1] = *(const v8bf*)(p + 16);
    return u.v;
}

// ---------------------------------------------------------------------------
// Kernel 1: transpose+convert the selected weight set to bf16 W^T[n][d]
// ---------------------------------------------------------------------------
__global__ __launch_bounds__(256) void prep_weights_kernel(
    const float* __restrict__ qWq, const float* __restrict__ qWk, const float* __restrict__ qWv,
    const float* __restrict__ dWq, const float* __restrict__ dWk, const float* __restrict__ dWv,
    const int* __restrict__ is_query,
    bf16_t* __restrict__ wqT, bf16_t* __restrict__ wkT, bf16_t* __restrict__ wvT)
{
    const int id = blockIdx.x * 256 + threadIdx.x;
    const int isq = *is_query;
    const float* Wq = isq ? qWq : dWq;
    const float* Wk = isq ? qWk : dWk;
    const float* Wv = isq ? qWv : dWv;
    const int QK = DIM * DIM;
    if (id < QK) {
        const int n = id / DIM, d = id - n * DIM;
        wqT[id] = (bf16_t)Wq[(size_t)d * DIM + n];
    } else if (id < 2 * QK) {
        const int i = id - QK;
        const int n = i / DIM, d = i - n * DIM;
        wkT[i] = (bf16_t)Wk[(size_t)d * DIM + n];
    } else if (id < 2 * QK + DOUTC * DIM) {
        const int i = id - 2 * QK;
        const int n = i / DIM, d = i - n * DIM;
        wvT[i] = (bf16_t)Wv[(size_t)d * DOUTC + n];
    }
}

// ---------------------------------------------------------------------------
// Kernel 2: projections.  C[16384, 1792] = emb @ [Wq | Wk | Wv] + bias
// Block = 8 waves, covers 128 rows x 64 cols; wave = 16x64 via 4 WMMA tiles,
// B-fragment loads software-pipelined one tile deep.
// Q stored pre-scaled by 1/sqrt(D) (bf16 row-major), K bf16 row-major,
// V bf16 TRANSPOSED [b][chan][s].
// ---------------------------------------------------------------------------
__global__ __launch_bounds__(256) void proj_kernel(
    const float* __restrict__ emb,
    const bf16_t* __restrict__ wqT, const bf16_t* __restrict__ wkT, const bf16_t* __restrict__ wvT,
    const float* __restrict__ q_bq, const float* __restrict__ q_bk, const float* __restrict__ q_bv,
    const float* __restrict__ d_bq, const float* __restrict__ d_bk, const float* __restrict__ d_bv,
    const int* __restrict__ is_query,
    bf16_t* __restrict__ Qo, bf16_t* __restrict__ Ko, bf16_t* __restrict__ Vt)
{
    const int lane = threadIdx.x & 31;
    const int w    = threadIdx.x >> 5;
    const int m0   = blockIdx.x * 128 + w * 16;
    const int ng   = blockIdx.y * 64;
    const int isq  = *is_query;

    const bf16_t* WT; const float* bias; int nloc, which;
    if (ng < DIM)           { WT = wqT; bias = isq ? q_bq : d_bq; nloc = ng;           which = 0; }
    else if (ng < 2 * DIM)  { WT = wkT; bias = isq ? q_bk : d_bk; nloc = ng - DIM;     which = 1; }
    else                    { WT = wvT; bias = isq ? q_bv : d_bv; nloc = ng - 2 * DIM; which = 2; }

    v8f acc[4] = {};
    for (int k0 = 0; k0 < DIM; k0 += 32) {
        const v16bf a = load_a_frag_f32(emb, DIM, m0, k0, lane);
        v16bf bpre = load_b_frag(WT, DIM, nloc, k0, lane);
#pragma unroll
        for (int j = 0; j < 4; ++j) {
            const v16bf bcur = bpre;
            if (j < 3) bpre = load_b_frag(WT, DIM, nloc + 16 * (j + 1), k0, lane);
            acc[j] = __builtin_amdgcn_wmma_f32_16x16x32_bf16(
                false, a, false, bcur, (short)0, acc[j], false, false);
        }
    }

    const float scale = (which == 0) ? INV_TEMP : 1.0f;
    const int rbase = (lane >> 4) * 8;
#pragma unroll
    for (int j = 0; j < 4; ++j) {
        const int col = nloc + 16 * j + (lane & 15);
        const float bv = bias[col];
#pragma unroll
        for (int r = 0; r < 8; ++r) {
            const int row = m0 + rbase + r;
            const float v = (acc[j][r] + bv) * scale;
            if (which == 0)      Qo[(size_t)row * DIM + col] = (bf16_t)v;
            else if (which == 1) Ko[(size_t)row * DIM + col] = (bf16_t)v;
            else {
                const int bb2 = row >> 10, s = row & (SEQ - 1);
                Vt[((size_t)bb2 * DOUTC + col) * SEQ + s] = (bf16_t)v;
            }
        }
    }
}

// ---------------------------------------------------------------------------
// Kernel 3: fused attention per (batch, 16 query rows).
// Q tile (16x768 bf16, 24KB) staged into LDS once per block via the gfx1250
// async global->LDS path (ASYNCcnt), then A-frags come from LDS while the
// VMEM pipe streams K-tile B-frags (pipelined one tile deep).
// Softmax: half-wave shfl_xor + cross-wave LDS reduction.
// attn (f32) -> d_out ; probs (bf16) -> LDS ; then P@V (K=1024) via WMMA.
// ---------------------------------------------------------------------------
__global__ __launch_bounds__(256) void attn_kernel(
    const bf16_t* __restrict__ Q, const bf16_t* __restrict__ Km,
    const bf16_t* __restrict__ Vt, const int* __restrict__ mask,
    float* __restrict__ out, float* __restrict__ attn)
{
    __shared__ __align__(32) bf16_t Qlds[16][DIM];   // 24 KB
    __shared__ __align__(32) bf16_t Plds[16][SEQ];   // 32 KB
    __shared__ float red[8][16];

    const int lane  = threadIdx.x & 31;
    const int w     = threadIdx.x >> 5;
    const int b     = blockIdx.y;
    const int q0    = blockIdx.x * 16;
    const int colc  = lane & 15;
    const int rbase = (lane >> 4) * 8;

    const bf16_t* Qb = Q + ((size_t)b * SEQ + q0) * DIM;
    const bf16_t* Kb = Km + (size_t)b * SEQ * DIM;

    // ---- stage Q tile to LDS: 16*768 bf16 = 1536 x 16B chunks, 6/thread ----
#if USE_ASYNC_LDS
    {
        bf16_t* dstbase = &Qlds[0][0];
#pragma unroll
        for (int j = 0; j < 6; ++j) {
            const int i = threadIdx.x + j * 256;          // chunk index
            __builtin_amdgcn_global_load_async_to_lds_b128(
                (as1_v4i*)(Qb + i * 8), (as3_v4i*)(dstbase + i * 8), 0, 0);
        }
        __builtin_amdgcn_s_wait_asynccnt(0);
    }
    __syncthreads();
#else
    {
        const v8bf* src = (const v8bf*)Qb;
        v8bf* dst = (v8bf*)&Qlds[0][0];
#pragma unroll
        for (int j = 0; j < 6; ++j) {
            const int i = threadIdx.x + j * 256;
            dst[i] = src[i];
        }
    }
    __syncthreads();
#endif

    // ---- scores: S[16,128] per wave, full D contraction ----
    v8f acc[8] = {};
    for (int k0 = 0; k0 < DIM; k0 += 32) {
        const v16bf a = load_a_frag_bf16(&Qlds[0][0], DIM, 0, k0, lane);
        v16bf bpre = load_b_frag(Kb, DIM, 128 * w, k0, lane);
#pragma unroll
        for (int t = 0; t < 8; ++t) {
            const v16bf bcur = bpre;
            if (t < 7) bpre = load_b_frag(Kb, DIM, 128 * w + 16 * (t + 1), k0, lane);
            acc[t] = __builtin_amdgcn_wmma_f32_16x16x32_bf16(
                false, a, false, bcur, (short)0, acc[t], false, false);
        }
    }

    // ---- mask + row max (per-lane partial over 8 key tiles) ----
    float rmax[8];
#pragma unroll
    for (int r = 0; r < 8; ++r) rmax[r] = -3.4e38f;
#pragma unroll
    for (int t = 0; t < 8; ++t) {
        const int col = 128 * w + 16 * t + colc;
        const int mk = mask[(size_t)b * SEQ + col];
#pragma unroll
        for (int r = 0; r < 8; ++r) {
            float sv = acc[t][r];
            sv = (mk == 0) ? -1e9f : sv;
            acc[t][r] = sv;
            rmax[r] = fmaxf(rmax[r], sv);
        }
    }
#pragma unroll
    for (int off = 1; off < 16; off <<= 1) {
#pragma unroll
        for (int r = 0; r < 8; ++r)
            rmax[r] = fmaxf(rmax[r], __shfl_xor(rmax[r], off, 32));
    }
    if (colc == 0) {
#pragma unroll
        for (int r = 0; r < 8; ++r) red[w][rbase + r] = rmax[r];
    }
    __syncthreads();
#pragma unroll
    for (int r = 0; r < 8; ++r) {
        float m = -3.4e38f;
#pragma unroll
        for (int w2 = 0; w2 < 8; ++w2) m = fmaxf(m, red[w2][rbase + r]);
        rmax[r] = m;
    }
    __syncthreads();

    // ---- exp + row sum ----
    float rsum[8];
#pragma unroll
    for (int r = 0; r < 8; ++r) rsum[r] = 0.0f;
#pragma unroll
    for (int t = 0; t < 8; ++t) {
#pragma unroll
        for (int r = 0; r < 8; ++r) {
            const float p = __expf(acc[t][r] - rmax[r]);
            acc[t][r] = p;
            rsum[r] += p;
        }
    }
#pragma unroll
    for (int off = 1; off < 16; off <<= 1) {
#pragma unroll
        for (int r = 0; r < 8; ++r)
            rsum[r] += __shfl_xor(rsum[r], off, 32);
    }
    if (colc == 0) {
#pragma unroll
        for (int r = 0; r < 8; ++r) red[w][rbase + r] = rsum[r];
    }
    __syncthreads();
    float rinv[8];
#pragma unroll
    for (int r = 0; r < 8; ++r) {
        float s = 0.0f;
#pragma unroll
        for (int w2 = 0; w2 < 8; ++w2) s += red[w2][rbase + r];
        rinv[r] = 1.0f / s;
    }

    // ---- normalize, emit attn (f32) and probs (bf16 -> LDS) ----
    float* attn_row = attn + ((size_t)b * SEQ + q0) * SEQ;
#pragma unroll
    for (int t = 0; t < 8; ++t) {
        const int col = 128 * w + 16 * t + colc;
#pragma unroll
        for (int r = 0; r < 8; ++r) {
            const int row = rbase + r;
            const float p = acc[t][r] * rinv[r];
            attn_row[(size_t)row * SEQ + col] = p;
            Plds[row][col] = (bf16_t)p;
        }
    }
    __syncthreads();

    // ---- out[16, 256] = P @ V ; wave w covers channels [32w, 32w+32) ----
    v8f oacc[2] = {};
    const bf16_t* Vb = Vt + (size_t)b * DOUTC * SEQ;
    const int n0 = 32 * w;
    for (int k0 = 0; k0 < SEQ; k0 += 32) {
        const v16bf a = load_a_frag_bf16(&Plds[0][0], SEQ, 0, k0, lane);
        v16bf bpre = load_b_frag(Vb, SEQ, n0, k0, lane);
#pragma unroll
        for (int j = 0; j < 2; ++j) {
            const v16bf bcur = bpre;
            if (j < 1) bpre = load_b_frag(Vb, SEQ, n0 + 16, k0, lane);
            oacc[j] = __builtin_amdgcn_wmma_f32_16x16x32_bf16(
                false, a, false, bcur, (short)0, oacc[j], false, false);
        }
    }
    float* out_row = out + ((size_t)b * SEQ + q0) * DOUTC;
#pragma unroll
    for (int j = 0; j < 2; ++j) {
        const int col = n0 + 16 * j + colc;
#pragma unroll
        for (int r = 0; r < 8; ++r)
            out_row[(size_t)(rbase + r) * DOUTC + col] = oacc[j][r];
    }
}

// ---------------------------------------------------------------------------
extern "C" void kernel_launch(void* const* d_in, const int* in_sizes, int n_in,
                              void* d_out, int out_size, void* d_ws, size_t ws_size,
                              hipStream_t stream) {
    const float* emb  = (const float*)d_in[0];
    const int*   mask = (const int*)d_in[1];
    const int*   isq  = (const int*)d_in[2];
    const float* qWq = (const float*)d_in[3];  const float* qbq = (const float*)d_in[4];
    const float* qWk = (const float*)d_in[5];  const float* qbk = (const float*)d_in[6];
    const float* qWv = (const float*)d_in[7];  const float* qbv = (const float*)d_in[8];
    const float* dWq = (const float*)d_in[9];  const float* dbq = (const float*)d_in[10];
    const float* dWk = (const float*)d_in[11]; const float* dbk = (const float*)d_in[12];
    const float* dWv = (const float*)d_in[13]; const float* dbv = (const float*)d_in[14];

    // Workspace layout (bf16 elements)
    bf16_t* ws  = (bf16_t*)d_ws;
    bf16_t* wqT = ws;                               // 768*768
    bf16_t* wkT = wqT + (size_t)DIM * DIM;          // 768*768
    bf16_t* wvT = wkT + (size_t)DIM * DIM;          // 256*768
    bf16_t* Qb  = wvT + (size_t)DOUTC * DIM;        // 16384*768 (pre-scaled)
    bf16_t* Kb  = Qb  + (size_t)BATCH * SEQ * DIM;  // 16384*768
    bf16_t* Vt  = Kb  + (size_t)BATCH * SEQ * DIM;  // 16*256*1024 (transposed)

    float* out  = (float*)d_out;                         // [B,S,DOUT]
    float* attn = out + (size_t)BATCH * SEQ * DOUTC;     // [B,S,S]

    // 1) weight transpose/convert (1,376,256 elements)
    const int prep_threads = 2 * DIM * DIM + DOUTC * DIM;
    prep_weights_kernel<<<(prep_threads + 255) / 256, 256, 0, stream>>>(
        qWq, qWk, qWv, dWq, dWk, dWv, isq, wqT, wkT, wvT);

    // 2) projections: M=16384 rows (128/block), N=1792 cols (64/block)
    proj_kernel<<<dim3((BATCH * SEQ) / 128, (2 * DIM + DOUTC) / 64), 256, 0, stream>>>(
        emb, wqT, wkT, wvT, qbq, qbk, qbv, dbq, dbk, dbv, isq, Qb, Kb, Vt);

    // 3) fused attention: one block per (16 query rows, batch)
    attn_kernel<<<dim3(SEQ / 16, BATCH), 256, 0, stream>>>(
        Qb, Kb, Vt, mask, out, attn);
}
